// Graph_Transformer_12317966205118
// MI455X (gfx1250) — matrix operand order repaired and checked by reading, not compiled, not run on hardware
//
#include <hip/hip_runtime.h>
#include <math.h>

// ---------------------------------------------------------------------------
// Graph Transformer block on MI455X (gfx1250): bf16 WMMA GEMMs + L2-resident
// edge softmax/aggregate with float atomics.
// ---------------------------------------------------------------------------

#define N_NODES 100000
#define E_EDGES 500000
#define DIM     128
#define HEADS   4
#define HIDDEN  512

typedef __bf16 bf16;
typedef __attribute__((ext_vector_type(16))) __bf16 v16bf;
typedef __attribute__((ext_vector_type(8)))  __bf16 v8bf;
typedef __attribute__((ext_vector_type(8)))  float  v8f;

// ---------------------------------------------------------------------------
// fp32 -> bf16 weight conversion (tiny, once per launch)
// ---------------------------------------------------------------------------
__global__ void f32_to_bf16_kernel(const float* __restrict__ src,
                                   bf16* __restrict__ dst, int n) {
    int i = blockIdx.x * 256 + threadIdx.x;
    if (i < n) dst[i] = (bf16)src[i];
}

// ---------------------------------------------------------------------------
// WMMA GEMM: Out[M x NOUT] = A[M x K] @ W[NOUT x K]^T + bias
// Each wave computes a 16x64 output patch = 4 WMMA tiles sharing one A frag
// per K-step (V_WMMA_F32_16X16X32_BF16, f32 accumulate). All 4 B fragments
// are loaded (one s_clause) before the 4 back-to-back WMMAs so matrix ops
// are not serialized behind individual loads.
//   A fragment (16x32 bf16): lane l -> row (l&15), elems [koff..koff+7] in
//     V0-3 and [16+koff..+7] in V4-7, koff=(l>>4)*8.
//   B fragment (32x16 bf16): lane l -> column (l&15) == W row, same K
//     layout as A: [koff..koff+7] and [16+koff..+7], koff=(l>>4)*8.
//   C/D: lane l holds column (l&15), rows (l>>4)*8 + r in acc[r].
// Block = 8 waves = (8 / (NOUT/64)) row strips x (NOUT/64) column groups.
// ---------------------------------------------------------------------------
template <int K, int NOUT, bool A_IS_BF16, bool RELU, bool OUT_BF16>
__global__ __launch_bounds__(256) void gemm_wmma_kernel(
    const void* __restrict__ Aptr, const bf16* __restrict__ W,
    const float* __restrict__ bias, void* __restrict__ Out, int M)
{
    constexpr int CGROUPS = NOUT / 64;      // column groups of 64 per row strip
    constexpr int RWAVES  = 8 / CGROUPS;    // row strips per block

    const int lane   = threadIdx.x & 31;
    const int wave   = threadIdx.x >> 5;    // 0..7
    const int rstrip = blockIdx.x * RWAVES + (wave / CGROUPS);
    const int row0   = rstrip * 16;
    if (row0 >= M) return;
    const int col0   = (wave % CGROUPS) * 64;

    const int mrow  = lane & 15;
    const int khalf = lane >> 4;            // 0 or 1
    const int koffA = khalf * 8;            // A: two 8-elem chunks
    const int koffB = khalf * 8;            // B: two 8-elem chunks (same as A)

    int arow = row0 + mrow;
    if (arow >= M) arow = M - 1;            // clamped load; stores are guarded

    const int nb = col0 + mrow;             // base output column / W row

    v8f acc[4];
    #pragma unroll
    for (int t = 0; t < 4; ++t) {
        const float b = bias[nb + 16 * t];
        #pragma unroll
        for (int i = 0; i < 8; ++i) acc[t][i] = b;
    }

    const bf16* __restrict__ Wbase = W + (size_t)nb * K + koffB;

    #pragma unroll
    for (int kt = 0; kt < K / 32; ++kt) {
        const int kb = kt * 32;
        // ---- all 4 B fragments first (single load clause) ----
        v16bf bfrag[4];
        #pragma unroll
        for (int t = 0; t < 4; ++t) {
            v8bf blo = *(const v8bf*)(Wbase + (size_t)(16 * t) * K + kb);
            v8bf bhi = *(const v8bf*)(Wbase + (size_t)(16 * t) * K + kb + 16);
            #pragma unroll
            for (int i = 0; i < 8; ++i) { bfrag[t][i] = blo[i]; bfrag[t][8 + i] = bhi[i]; }
        }
        // ---- A fragment (shared by the 4 column tiles) ----
        v16bf afrag;
        if (A_IS_BF16) {
            const bf16* __restrict__ Arow = (const bf16*)Aptr + (size_t)arow * K;
            v8bf lo = *(const v8bf*)(Arow + kb + koffA);
            v8bf hi = *(const v8bf*)(Arow + kb + 16 + koffA);
            #pragma unroll
            for (int i = 0; i < 8; ++i) { afrag[i] = lo[i]; afrag[8 + i] = hi[i]; }
        } else {
            const float* __restrict__ Arow = (const float*)Aptr + (size_t)arow * K;
            float4 a0 = *(const float4*)(Arow + kb + koffA);
            float4 a1 = *(const float4*)(Arow + kb + koffA + 4);
            float4 a2 = *(const float4*)(Arow + kb + 16 + koffA);
            float4 a3 = *(const float4*)(Arow + kb + 16 + koffA + 4);
            afrag[0] = (bf16)a0.x;  afrag[1] = (bf16)a0.y;
            afrag[2] = (bf16)a0.z;  afrag[3] = (bf16)a0.w;
            afrag[4] = (bf16)a1.x;  afrag[5] = (bf16)a1.y;
            afrag[6] = (bf16)a1.z;  afrag[7] = (bf16)a1.w;
            afrag[8]  = (bf16)a2.x; afrag[9]  = (bf16)a2.y;
            afrag[10] = (bf16)a2.z; afrag[11] = (bf16)a2.w;
            afrag[12] = (bf16)a3.x; afrag[13] = (bf16)a3.y;
            afrag[14] = (bf16)a3.z; afrag[15] = (bf16)a3.w;
        }
        // ---- 4 back-to-back WMMAs ----
        #pragma unroll
        for (int t = 0; t < 4; ++t) {
            acc[t] = __builtin_amdgcn_wmma_f32_16x16x32_bf16(
                /*neg_a=*/false, afrag, /*neg_b=*/false, bfrag[t],
                /*c_mod=*/(short)0, acc[t], /*reuse_a=*/false, /*reuse_b=*/false);
        }
    }

    // ---- store: lane holds column n, rows khalf*8 + r ----
    const int mbase = khalf * 8;
    #pragma unroll
    for (int t = 0; t < 4; ++t) {
        const int n = nb + 16 * t;
        #pragma unroll
        for (int r = 0; r < 8; ++r) {
            const int m = row0 + mbase + r;
            if (m < M) {
                float val = acc[t][r];
                if (RELU) val = fmaxf(val, 0.0f);
                if (OUT_BF16)
                    ((bf16*)Out)[(size_t)m * NOUT + n] = (bf16)val;
                else
                    ((float*)Out)[(size_t)m * NOUT + n] = val;
            }
        }
    }
}

// ---------------------------------------------------------------------------
// m = -inf, z = 0 init
// ---------------------------------------------------------------------------
__global__ void init_mz_kernel(float* __restrict__ m, float* __restrict__ z, int n) {
    int i = blockIdx.x * 256 + threadIdx.x;
    if (i < n) { m[i] = -INFINITY; z[i] = 0.0f; }
}

// signed/unsigned split atomic float max (addr pre-initialized to -inf)
__device__ __forceinline__ void atomicMaxFloat(float* addr, float v) {
    if (v >= 0.0f) atomicMax((int*)addr, __float_as_int(v));
    else           atomicMin((unsigned int*)addr, __float_as_uint(v));
}

// ---------------------------------------------------------------------------
// Pass 1: alpha[e,h] = dot(q[dst,h,:], k[src,h,:]) / sqrt(C); segment max.
// One wave per edge; lane handles 4 channels; heads reduce over 8-lane groups.
// q/k working set (102 MB) is L2-resident on the 192 MB L2.
// ---------------------------------------------------------------------------
__global__ __launch_bounds__(256) void edge_alpha_max_kernel(
    const long long* __restrict__ ei, const float* __restrict__ q,
    const float* __restrict__ k, float* __restrict__ alpha,
    float* __restrict__ mbuf, int E)
{
    const int lane = threadIdx.x & 31;
    const int e = blockIdx.x * 8 + (threadIdx.x >> 5);
    if (e >= E) return;
    const int src = (int)ei[e];
    const int dst = (int)ei[(size_t)E + e];
    const int c4 = lane * 4;
    float4 qv = *(const float4*)(q + (size_t)dst * DIM + c4);
    float4 kv = *(const float4*)(k + (size_t)src * DIM + c4);
    float s = qv.x * kv.x + qv.y * kv.y + qv.z * kv.z + qv.w * kv.w;
    s += __shfl_xor(s, 1);
    s += __shfl_xor(s, 2);
    s += __shfl_xor(s, 4);                   // per-head sum in lanes 8h
    if ((lane & 7) == 0) {
        const int h = lane >> 3;
        const float a = s * 0.17677669529663687f;    // 1/sqrt(32)
        alpha[(size_t)e * HEADS + h] = a;
        atomicMaxFloat(mbuf + (size_t)dst * HEADS + h, a);
    }
}

// ---------------------------------------------------------------------------
// Pass 2: e = exp(alpha - m[dst]); z[dst] += e  (thread per edge*head)
// ---------------------------------------------------------------------------
__global__ void edge_exp_sum_kernel(const long long* __restrict__ ei,
                                    float* __restrict__ alpha,
                                    const float* __restrict__ m,
                                    float* __restrict__ z, int E)
{
    int i = blockIdx.x * 256 + threadIdx.x;
    if (i >= E * HEADS) return;
    const int e = i >> 2;
    const int h = i & 3;
    const int dst = (int)ei[(size_t)E + e];
    const float ex = expf(alpha[i] - m[(size_t)dst * HEADS + h]);
    alpha[i] = ex;
    atomicAdd(z + (size_t)dst * HEADS + h, ex);
}

// ---------------------------------------------------------------------------
// Pass 3: conv[dst] += (e/z[dst]) * v[src]   (conv pre-loaded with skip term)
// One wave per edge, 4 float atomic adds per lane; v+conv L2-resident.
// ---------------------------------------------------------------------------
__global__ __launch_bounds__(256) void edge_aggregate_kernel(
    const long long* __restrict__ ei, const float* __restrict__ alpha,
    const float* __restrict__ z, const float* __restrict__ v,
    float* __restrict__ conv, int E)
{
    const int lane = threadIdx.x & 31;
    const int e = blockIdx.x * 8 + (threadIdx.x >> 5);
    if (e >= E) return;
    const int src = (int)ei[e];
    const int dst = (int)ei[(size_t)E + e];
    const int h = lane >> 3;
    const float a = alpha[(size_t)e * HEADS + h] / z[(size_t)dst * HEADS + h];
    const int c4 = lane * 4;
    float4 vv = *(const float4*)(v + (size_t)src * DIM + c4);
    float* out = conv + (size_t)dst * DIM + c4;
    atomicAdd(out + 0, a * vv.x);
    atomicAdd(out + 1, a * vv.y);
    atomicAdd(out + 2, a * vv.z);
    atomicAdd(out + 3, a * vv.w);
}

// ---------------------------------------------------------------------------
// out = LayerNorm(a + b) * w + bias   (one wave per 128-wide row)
// ---------------------------------------------------------------------------
__global__ __launch_bounds__(256) void ln_residual_kernel(
    const float* __restrict__ a, const float* __restrict__ b,
    const float* __restrict__ w, const float* __restrict__ bias,
    float* __restrict__ out, int M)
{
    const int lane = threadIdx.x & 31;
    const int row = blockIdx.x * 8 + (threadIdx.x >> 5);
    if (row >= M) return;
    const size_t base = (size_t)row * DIM;
    float4 av = *(const float4*)(a + base + lane * 4);
    float4 bv = *(const float4*)(b + base + lane * 4);
    float y0 = av.x + bv.x, y1 = av.y + bv.y, y2 = av.z + bv.z, y3 = av.w + bv.w;
    float s = y0 + y1 + y2 + y3;
    #pragma unroll
    for (int msk = 1; msk < 32; msk <<= 1) s += __shfl_xor(s, msk);
    const float mu = s * (1.0f / DIM);
    float d0 = y0 - mu, d1 = y1 - mu, d2 = y2 - mu, d3 = y3 - mu;
    float vs = d0 * d0 + d1 * d1 + d2 * d2 + d3 * d3;
    #pragma unroll
    for (int msk = 1; msk < 32; msk <<= 1) vs += __shfl_xor(vs, msk);
    const float inv = rsqrtf(vs * (1.0f / DIM) + 1e-5f);
    float4 wv = *(const float4*)(w + lane * 4);
    float4 gv = *(const float4*)(bias + lane * 4);
    float4 o;
    o.x = d0 * inv * wv.x + gv.x;
    o.y = d1 * inv * wv.y + gv.y;
    o.z = d2 * inv * wv.z + gv.z;
    o.w = d3 * inv * wv.w + gv.w;
    *(float4*)(out + base + lane * 4) = o;
}

// ---------------------------------------------------------------------------
// Host launch
// ---------------------------------------------------------------------------
extern "C" void kernel_launch(void* const* d_in, const int* in_sizes, int n_in,
                              void* d_out, int out_size, void* d_ws, size_t ws_size,
                              hipStream_t stream) {
    (void)in_sizes; (void)n_in; (void)out_size; (void)ws_size;

    const float*     x     = (const float*)d_in[0];
    const long long* ei    = (const long long*)d_in[1];
    const float*     Wq    = (const float*)d_in[2];
    const float*     bq    = (const float*)d_in[3];
    const float*     Wk    = (const float*)d_in[4];
    const float*     bk    = (const float*)d_in[5];
    const float*     Wv    = (const float*)d_in[6];
    const float*     bv    = (const float*)d_in[7];
    const float*     Wskip = (const float*)d_in[8];
    const float*     bskip = (const float*)d_in[9];
    const float*     Wout  = (const float*)d_in[10];
    const float*     bout  = (const float*)d_in[11];
    const float*     ln1_w = (const float*)d_in[12];
    const float*     ln1_b = (const float*)d_in[13];
    const float*     ln2_w = (const float*)d_in[14];
    const float*     ln2_b = (const float*)d_in[15];
    const float*     W1    = (const float*)d_in[16];
    const float*     b1    = (const float*)d_in[17];
    const float*     W2    = (const float*)d_in[18];
    const float*     b2    = (const float*)d_in[19];
    float* out = (float*)d_out;

    // ---- workspace layout (256B aligned) ----
    char* ws = (char*)d_ws;
    const size_t szND   = (size_t)N_NODES * DIM * sizeof(float);   // 51.2 MB (mult of 256)
    const size_t szA    = (size_t)E_EDGES * HEADS * sizeof(float); // 8 MB
    const size_t szNH   = (size_t)N_NODES * HEADS * sizeof(float); // 1.6 MB
    const size_t szWb   = (size_t)DIM * DIM * sizeof(bf16);        // 32 KB
    const size_t szW1b  = (size_t)HIDDEN * DIM * sizeof(bf16);     // 128 KB

    size_t off = 0;
    float* qbuf  = (float*)(ws + off); off += szND;   // q  } adjacent: reused as
    float* kbuf  = (float*)(ws + off); off += szND;   // k  } bf16 FFN hidden buf
    float* vbuf  = (float*)(ws + off); off += szND;
    float* conv  = (float*)(ws + off); off += szND;   // skip + aggregated messages
    float* out1  = (float*)(ws + off); off += szND;   // also reused for out3
    float* out2  = (float*)(ws + off); off += szND;
    float* alpha = (float*)(ws + off); off += szA;
    float* mbuf  = (float*)(ws + off); off += szNH;
    float* zbuf  = (float*)(ws + off); off += szNH;
    bf16* wq_b   = (bf16*)(ws + off); off += szWb;
    bf16* wk_b   = (bf16*)(ws + off); off += szWb;
    bf16* wv_b   = (bf16*)(ws + off); off += szWb;
    bf16* ws_b   = (bf16*)(ws + off); off += szWb;
    bf16* wo_b   = (bf16*)(ws + off); off += szWb;
    bf16* w1_b   = (bf16*)(ws + off); off += szW1b;
    bf16* w2_b   = (bf16*)(ws + off); off += szW1b;
    bf16* h_bf   = (bf16*)qbuf;   // N*512 bf16 == q||k region, q/k dead by then
    float* out3  = out1;          // out1 dead after first LayerNorm

    const dim3 blk(256);

    // ---- weight conversion to bf16 ----
    f32_to_bf16_kernel<<<dim3(64),  blk, 0, stream>>>(Wq,    wq_b, DIM * DIM);
    f32_to_bf16_kernel<<<dim3(64),  blk, 0, stream>>>(Wk,    wk_b, DIM * DIM);
    f32_to_bf16_kernel<<<dim3(64),  blk, 0, stream>>>(Wv,    wv_b, DIM * DIM);
    f32_to_bf16_kernel<<<dim3(64),  blk, 0, stream>>>(Wskip, ws_b, DIM * DIM);
    f32_to_bf16_kernel<<<dim3(64),  blk, 0, stream>>>(Wout,  wo_b, DIM * DIM);
    f32_to_bf16_kernel<<<dim3(256), blk, 0, stream>>>(W1,    w1_b, HIDDEN * DIM);
    f32_to_bf16_kernel<<<dim3(256), blk, 0, stream>>>(W2,    w2_b, DIM * HIDDEN);

    // NOUT=128: block = 4 row strips x 2 col groups -> 64 rows per block
    const dim3 g128((N_NODES + 63) / 64, 1);
    // NOUT=512: block = 1 row strip x 8 col groups -> 16 rows per block
    const dim3 g512((N_NODES + 15) / 16, 1);

    // ---- q / k / v ; skip written straight into conv accumulator ----
    gemm_wmma_kernel<128, 128, false, false, false><<<g128, blk, 0, stream>>>(x, wq_b, bq,    qbuf, N_NODES);
    gemm_wmma_kernel<128, 128, false, false, false><<<g128, blk, 0, stream>>>(x, wk_b, bk,    kbuf, N_NODES);
    gemm_wmma_kernel<128, 128, false, false, false><<<g128, blk, 0, stream>>>(x, wv_b, bv,    vbuf, N_NODES);
    gemm_wmma_kernel<128, 128, false, false, false><<<g128, blk, 0, stream>>>(x, ws_b, bskip, conv, N_NODES);

    // ---- segment softmax + aggregate ----
    init_mz_kernel<<<dim3((N_NODES * HEADS + 255) / 256), blk, 0, stream>>>(mbuf, zbuf, N_NODES * HEADS);
    edge_alpha_max_kernel<<<dim3((E_EDGES + 7) / 8), blk, 0, stream>>>(ei, qbuf, kbuf, alpha, mbuf, E_EDGES);
    edge_exp_sum_kernel<<<dim3((E_EDGES * HEADS + 255) / 256), blk, 0, stream>>>(ei, alpha, mbuf, zbuf, E_EDGES);
    edge_aggregate_kernel<<<dim3((E_EDGES + 7) / 8), blk, 0, stream>>>(ei, alpha, zbuf, vbuf, conv, E_EDGES);

    // ---- out1 = conv @ Wout.T + bout ; out2 = LN(out1 + x) ----
    gemm_wmma_kernel<128, 128, false, false, false><<<g128, blk, 0, stream>>>(conv, wo_b, bout, out1, N_NODES);
    ln_residual_kernel<<<dim3((N_NODES + 7) / 8), blk, 0, stream>>>(out1, x, ln1_w, ln1_b, out2, N_NODES);

    // ---- FFN: h = relu(out2 @ W1.T + b1) (bf16) ; out3 = h @ W2.T + b2 ----
    gemm_wmma_kernel<128, 512, false, true,  true ><<<g512, blk, 0, stream>>>(out2, w1_b, b1, h_bf, N_NODES);
    gemm_wmma_kernel<512, 128, true,  false, false><<<g128, blk, 0, stream>>>(h_bf, w2_b, b2, out3, N_NODES);

    // ---- out = LN(out3 + out2) ----
    ln_residual_kernel<<<dim3((N_NODES + 7) / 8), blk, 0, stream>>>(out3, out2, ln2_w, ln2_b, out, N_NODES);
}
